// DynamicRNNEncoder_82970178224424
// MI455X (gfx1250) — matrix operand order repaired and checked by reading, not compiled, not run on hardware
//
#include <hip/hip_runtime.h>
#include <cstdint>
#include <cstddef>

namespace {

constexpr int Bn = 64, Tn = 256, Dn = 512, Hn = 1024, AGGn = 2048, Zn = 256;

typedef __attribute__((ext_vector_type(16))) __bf16 v16bf;
typedef __attribute__((ext_vector_type(8)))  float  v8f;

__device__ __forceinline__ unsigned short f32_to_bf16_bits(float f) {
  unsigned int u = __builtin_bit_cast(unsigned int, f);
  u += 0x7FFFu + ((u >> 16) & 1u);           // round-to-nearest-even
  return (unsigned short)(u >> 16);
}
__device__ __forceinline__ float bf16_bits_to_f32(unsigned short s) {
  unsigned int u = ((unsigned int)s) << 16;
  return __builtin_bit_cast(float, u);
}

union FragU { v16bf v; uint4 q[2]; };

// A fragment: 16x32 bf16, row-major source [M x K], lda in elements.
// Layout: lanes 0-15 row=lane, K = 0..15; lanes 16-31 row=lane-16, K = 16..31.
__device__ __forceinline__ v16bf load_frag_a(const unsigned short* __restrict__ A,
                                             int lda, int rowBase, int k0) {
  int lane = threadIdx.x & 31;
  int half = lane >> 4, idx = lane & 15;
  const unsigned short* r = A + (size_t)(rowBase + idx) * (size_t)lda + k0 + 16 * half;
  FragU f;
  f.q[0] = *reinterpret_cast<const uint4*>(r);       // K base+0..7
  f.q[1] = *reinterpret_cast<const uint4*>(r + 8);   // K base+8..15
  return f.v;
}

// B fragment from pre-transposed weights Bt [N x K] row-major (ldb = K).
// lane = output column; lanes 0-15 hold K 0..15, lanes 16-31 hold K 16..31 (contig 32B).
__device__ __forceinline__ v16bf load_frag_bt(const unsigned short* __restrict__ Bt,
                                              int ldb, int colBase, int k0) {
  int lane = threadIdx.x & 31;
  int half = lane >> 4, idx = lane & 15;
  const unsigned short* r = Bt + (size_t)(colBase + idx) * (size_t)ldb + k0 + 16 * half;
  FragU f;
  f.q[0] = *reinterpret_cast<const uint4*>(r);
  f.q[1] = *reinterpret_cast<const uint4*>(r + 8);
  return f.v;
}

// Same layout as load_frag_bt but reading a [16 x ldb] tile staged in LDS.
__device__ __forceinline__ v16bf load_frag_b_lds(const unsigned short* lds,
                                                 int ldb, int k0) {
  int lane = threadIdx.x & 31;
  int half = lane >> 4, idx = lane & 15;
  const unsigned short* r = lds + idx * ldb + k0 + 16 * half;
  FragU f;
  f.q[0] = *reinterpret_cast<const uint4*>(r);
  f.q[1] = *reinterpret_cast<const uint4*>(r + 8);
  return f.v;
}

__device__ __forceinline__ v8f wmma_bf16(v16bf a, v16bf b, v8f c) {
  return __builtin_amdgcn_wmma_f32_16x16x32_bf16(false, a, false, b, (short)0, c,
                                                 false, false);
}

__device__ __forceinline__ float sigmoidf_fast(float x) {
  return 1.0f / (1.0f + __expf(-x));
}

// Async copy of 16 bytes global -> LDS (ASYNCcnt-tracked, no VGPR round trip).
__device__ __forceinline__ void async_copy_b128(unsigned lds_byte_addr,
                                                const void* gptr) {
  unsigned long long ga = (unsigned long long)(uintptr_t)gptr;
  asm volatile("global_load_async_to_lds_b128 %0, %1, off"
               :: "v"(lds_byte_addr), "v"(ga) : "memory");
}
__device__ __forceinline__ void wait_asynccnt0() {
  asm volatile("s_wait_asynccnt 0x0" ::: "memory");
}

} // namespace

// ---------------- conversion kernels ----------------

__global__ void cast_bf16_kernel(const float* __restrict__ in,
                                 unsigned short* __restrict__ out, int n) {
  for (int i = blockIdx.x * blockDim.x + threadIdx.x; i < n;
       i += gridDim.x * blockDim.x)
    out[i] = f32_to_bf16_bits(in[i]);
}

// in: [K x N] row-major f32  ->  out: [N x K] row-major bf16
__global__ void transpose_cast_kernel(const float* __restrict__ in,
                                      unsigned short* __restrict__ out, int K, int N) {
  int total = K * N;
  for (int i = blockIdx.x * blockDim.x + threadIdx.x; i < total;
       i += gridDim.x * blockDim.x) {
    int k = i % K, n = i / K;
    out[i] = f32_to_bf16_bits(in[(size_t)k * N + n]);
  }
}

// ---------------- recurrence ----------------

// state[b,h..h+7] = sum_{j<step} cond[b,step,j] * buf[j,b,h..h+7]
// (B*H/8 threads; b128 streaming of buf)
__global__ void state_kernel(const float* __restrict__ cond,
                             const unsigned short* __restrict__ buf,
                             float* __restrict__ state_f,
                             unsigned short* __restrict__ state_b, int step) {
  int gid = blockIdx.x * blockDim.x + threadIdx.x;   // Bn * Hn/8 threads
  int b  = gid >> 7;                                 // Hn/8 = 128 chunks per batch
  int hc = (gid & 127) * 8;
  const float* crow = cond + (size_t)b * Tn * Tn + (size_t)step * Tn;
  float acc[8] = {0.f, 0.f, 0.f, 0.f, 0.f, 0.f, 0.f, 0.f};
  for (int j = 0; j < step; ++j) {
    float c = crow[j];
    uint4 q = *reinterpret_cast<const uint4*>(buf + ((size_t)j * Bn + b) * Hn + hc);
    const unsigned short* u = reinterpret_cast<const unsigned short*>(&q);
#pragma unroll
    for (int e = 0; e < 8; ++e) acc[e] += c * bf16_bits_to_f32(u[e]);
  }
  size_t o = (size_t)b * Hn + hc;
#pragma unroll
  for (int e = 0; e < 8; ++e) state_f[o + e] = acc[e];
  unsigned short sb[8];
#pragma unroll
  for (int e = 0; e < 8; ++e) sb[e] = f32_to_bf16_bits(acc[e]);
  *reinterpret_cast<uint4*>(state_b + o) = *reinterpret_cast<const uint4*>(sb);
}

// One wave per 16x16 output tile of the hidden state. Fuses the concatenated-K
// GEMM ([state|x] against [U;W]) for all three gate column-blocks, plus the
// GRU nonlinearity (keras reset_after=True). Weights stream from L2 (9 MB/step,
// no intra-workgroup reuse -> no LDS staging here).
__global__ void __launch_bounds__(32)
gru_step_kernel(const unsigned short* __restrict__ xb,   // [B,T,D] bf16
                const unsigned short* __restrict__ Wt,   // [3H x D] bf16 (W^T)
                const unsigned short* __restrict__ Ut,   // [3H x H] bf16 (U^T)
                const float* __restrict__ bias,          // [2,3H] f32
                const float* __restrict__ state_f,       // [B,H] f32
                const unsigned short* __restrict__ state_b, // [B,H] bf16
                unsigned short* __restrict__ buf,        // [T,B,H] bf16
                int step) {
  const int rowBase = blockIdx.x * 16;  // batch tile
  const int colBase = blockIdx.y * 16;  // hidden-col tile
  v8f accZ  = {0.f,0.f,0.f,0.f,0.f,0.f,0.f,0.f};
  v8f accR  = accZ;
  v8f accHH = accZ;   // state @ U_h   (scaled by r later)
  v8f accXH = accZ;   // x @ W_h

  // K over H: state @ U for all three gate blocks (A-frag reused x3)
  for (int k0 = 0; k0 < Hn; k0 += 32) {
    v16bf a = load_frag_a(state_b, Hn, rowBase, k0);
    accZ  = wmma_bf16(a, load_frag_bt(Ut, Hn, colBase,          k0), accZ);
    accR  = wmma_bf16(a, load_frag_bt(Ut, Hn, colBase + Hn,     k0), accR);
    accHH = wmma_bf16(a, load_frag_bt(Ut, Hn, colBase + 2 * Hn, k0), accHH);
  }
  // K over D: x @ W (z/r fused into accZ/accR, h kept separate)
  const unsigned short* xrow = xb + (size_t)step * Dn;  // row stride = T*D
  for (int k0 = 0; k0 < Dn; k0 += 32) {
    v16bf a = load_frag_a(xrow, Tn * Dn, rowBase, k0);
    accZ  = wmma_bf16(a, load_frag_bt(Wt, Dn, colBase,          k0), accZ);
    accR  = wmma_bf16(a, load_frag_bt(Wt, Dn, colBase + Hn,     k0), accR);
    accXH = wmma_bf16(a, load_frag_bt(Wt, Dn, colBase + 2 * Hn, k0), accXH);
  }

  const int lane = threadIdx.x & 31, half = lane >> 4, idx = lane & 15;
  const int hg = colBase + idx;
  const float b0z = bias[hg],           b1z = bias[3 * Hn + hg];
  const float b0r = bias[Hn + hg],      b1r = bias[4 * Hn + hg];
  const float b0h = bias[2 * Hn + hg],  b1h = bias[5 * Hn + hg];
#pragma unroll
  for (int r = 0; r < 8; ++r) {
    const int bg = rowBase + r + 8 * half;      // C/D layout: VGPR r -> M = r + 8*half
    float z  = sigmoidf_fast(accZ[r] + b0z + b1z);
    float rr = sigmoidf_fast(accR[r] + b0r + b1r);
    float c  = tanhf(accXH[r] + b0h + rr * (accHH[r] + b1h));
    float hp = state_f[(size_t)bg * Hn + hg];
    float hn = z * hp + (1.0f - z) * c;
    buf[((size_t)step * Bn + bg) * Hn + hg] = f32_to_bf16_bits(hn);
  }
}

// ---------------- DeepSet ----------------

// 4 waves per block; each block owns one 16-wide AGG column tile and stages its
// loop-invariant Wit tile (16x1024 bf16 = 32 KB) into LDS ONCE via async-to-LDS,
// then loops all T timesteps (one batch tile per wave), relu-per-t, register
// accumulation of the time-sum. B fragments come from LDS (ds_load_b128 x2).
__global__ void __launch_bounds__(128)
deepset_inner_kernel(const unsigned short* __restrict__ buf,   // [T,B,H] bf16
                     const unsigned short* __restrict__ Wit,   // [AGG x H] bf16
                     const float* __restrict__ bi,
                     float* __restrict__ agg_f,
                     unsigned short* __restrict__ agg_b) {
  __shared__ unsigned short lds_w[16 * Hn];        // 32 KB column tile of Wit
  const int colBase = blockIdx.x * 16;
  const int rowBase = (threadIdx.x >> 5) * 16;     // one batch tile per wave
  const int lane = threadIdx.x & 31, half = lane >> 4, idx = lane & 15;

  // Cooperative async stage: 16384 bf16 = 32 KB, 128 threads x 16 iters x 16 B.
  {
    const unsigned short* src = Wit + (size_t)colBase * Hn;  // contiguous 16 rows
    const unsigned lds_base = (unsigned)(uintptr_t)&lds_w[0];
    const int tid = threadIdx.x;
#pragma unroll
    for (int it = 0; it < 16; ++it) {
      unsigned eo = (unsigned)(it * 128 + tid) * 8;          // element offset
      async_copy_b128(lds_base + eo * 2, src + eo);
    }
    wait_asynccnt0();
    __syncthreads();
  }

  const float bias = bi[colBase + idx];
  v8f aggAcc = {0.f,0.f,0.f,0.f,0.f,0.f,0.f,0.f};
  for (int t = 0; t < Tn; ++t) {
    v8f acc = {0.f,0.f,0.f,0.f,0.f,0.f,0.f,0.f};
    const unsigned short* At = buf + (size_t)t * Bn * Hn;
    for (int k0 = 0; k0 < Hn; k0 += 32)
      acc = wmma_bf16(load_frag_a(At, Hn, rowBase, k0),
                      load_frag_b_lds(lds_w, Hn, k0), acc);
#pragma unroll
    for (int r = 0; r < 8; ++r) {
      float v = acc[r] + bias;
      aggAcc[r] += (v > 0.0f) ? v : 0.0f;
    }
  }
#pragma unroll
  for (int r = 0; r < 8; ++r) {
    const int bg = rowBase + r + 8 * half, cg = colBase + idx;
    agg_f[(size_t)bg * AGGn + cg] = aggAcc[r];
    agg_b[(size_t)bg * AGGn + cg] = f32_to_bf16_bits(aggAcc[r]);
  }
}

__global__ void __launch_bounds__(32)
deepset_outer_kernel(const unsigned short* __restrict__ agg_b,  // [B x AGG] bf16
                     const unsigned short* __restrict__ Wot,    // [Z x AGG] bf16
                     const float* __restrict__ bo,
                     float* __restrict__ out) {               // [B x Z] f32
  const int rowBase = blockIdx.x * 16;
  const int colBase = blockIdx.y * 16;
  const int lane = threadIdx.x & 31, half = lane >> 4, idx = lane & 15;
  v8f acc = {0.f,0.f,0.f,0.f,0.f,0.f,0.f,0.f};
  for (int k0 = 0; k0 < AGGn; k0 += 32)
    acc = wmma_bf16(load_frag_a(agg_b, AGGn, rowBase, k0),
                    load_frag_bt(Wot, AGGn, colBase, k0), acc);
#pragma unroll
  for (int r = 0; r < 8; ++r) {
    const int bg = rowBase + r + 8 * half, cg = colBase + idx;
    float v = acc[r] + bo[cg];
    out[(size_t)bg * Zn + cg] = (v > 0.0f) ? v : 0.0f;
  }
}

// ---------------- launch ----------------

extern "C" void kernel_launch(void* const* d_in, const int* in_sizes, int n_in,
                              void* d_out, int out_size, void* d_ws, size_t ws_size,
                              hipStream_t stream) {
  (void)in_sizes; (void)n_in; (void)out_size; (void)ws_size;
  const float* input = (const float*)d_in[0];  // [B,T,D]
  const float* cond  = (const float*)d_in[1];  // [B,T,T]
  const float* W     = (const float*)d_in[2];  // [D,3H]
  const float* U     = (const float*)d_in[3];  // [H,3H]
  const float* bias  = (const float*)d_in[4];  // [2,3H]
  const float* Wi    = (const float*)d_in[5];  // [H,AGG]
  const float* bi    = (const float*)d_in[6];  // [AGG]
  const float* Wo    = (const float*)d_in[7];  // [AGG,Z]
  const float* bo    = (const float*)d_in[8];  // [Z]
  float* out = (float*)d_out;

  char* p = (char*)d_ws;
  auto bump = [&](size_t bytes) -> char* {
    char* r = p;
    p += (bytes + 255) & ~size_t(255);
    return r;
  };
  unsigned short* xb      = (unsigned short*)bump((size_t)Bn * Tn * Dn * 2);
  unsigned short* Wt      = (unsigned short*)bump((size_t)3 * Hn * Dn * 2);
  unsigned short* Ut      = (unsigned short*)bump((size_t)3 * Hn * Hn * 2);
  unsigned short* Wit     = (unsigned short*)bump((size_t)AGGn * Hn * 2);
  unsigned short* Wot     = (unsigned short*)bump((size_t)Zn * AGGn * 2);
  float*          state_f = (float*)bump((size_t)Bn * Hn * 4);
  unsigned short* state_b = (unsigned short*)bump((size_t)Bn * Hn * 2);
  unsigned short* buf     = (unsigned short*)bump((size_t)Tn * Bn * Hn * 2);
  float*          agg_f   = (float*)bump((size_t)Bn * AGGn * 4);
  unsigned short* agg_b   = (unsigned short*)bump((size_t)Bn * AGGn * 2);

  // One-time bf16 conversions (weights transposed to [N x K] for contiguous
  // B-fragment loads).
  cast_bf16_kernel<<<1024, 256, 0, stream>>>(input, xb, Bn * Tn * Dn);
  transpose_cast_kernel<<<1024, 256, 0, stream>>>(W,  Wt,  Dn,   3 * Hn);
  transpose_cast_kernel<<<1024, 256, 0, stream>>>(U,  Ut,  Hn,   3 * Hn);
  transpose_cast_kernel<<<1024, 256, 0, stream>>>(Wi, Wit, Hn,   AGGn);
  transpose_cast_kernel<<<1024, 256, 0, stream>>>(Wo, Wot, AGGn, Zn);

  // Sequential scan: weighted-state reduction + fused WMMA GRU step.
  for (int i = 0; i < Tn; ++i) {
    state_kernel<<<(Bn * Hn / 8) / 256, 256, 0, stream>>>(cond, buf, state_f,
                                                          state_b, i);
    gru_step_kernel<<<dim3(Bn / 16, Hn / 16), 32, 0, stream>>>(
        xb, Wt, Ut, bias, state_f, state_b, buf, i);
  }

  // DeepSet: fused relu+time-sum GEMM with async-LDS-staged weights, then the
  // tiny output GEMM.
  deepset_inner_kernel<<<AGGn / 16, 128, 0, stream>>>(buf, Wit, bi, agg_f, agg_b);
  deepset_outer_kernel<<<dim3(Bn / 16, Zn / 16), 32, 0, stream>>>(agg_b, Wot, bo, out);
}
